// SubBlockConv2d_8847632630078
// MI455X (gfx1250) — matrix operand order repaired and checked by reading, not compiled, not run on hardware
//
#include <hip/hip_runtime.h>

// ---- CDNA5 (gfx1250, wave32) bf16 WMMA implementation ----
typedef __attribute__((ext_vector_type(16))) __bf16 v16bf;
typedef __attribute__((ext_vector_type(8)))  __bf16 v8bf;
typedef __attribute__((ext_vector_type(4)))  __bf16 v4bf;
typedef __attribute__((ext_vector_type(8)))  float  v8f;

#define IN_C     192
#define OUT_CONV 192               // 3 * 8 * 8
#define NB       16
#define IMH      80
#define IMW      120
#define NPIX     (IMH * IMW)       // 9600 pixels per batch image
#define TILE_PX  128               // pixels per workgroup
#define TILES_PB (NPIX / TILE_PX)  // 75 (exact)
#define LSTRIDE  200               // padded channel stride in LDS (elements)

__device__ __forceinline__ __bf16 f2bf_rne(float f) {
  unsigned u = __builtin_bit_cast(unsigned, f);
  u += 0x7FFFu + ((u >> 16) & 1u);           // round-to-nearest-even
  unsigned short s = (unsigned short)(u >> 16);
  return __builtin_bit_cast(__bf16, s);
}

__global__ __launch_bounds__(384)
void subblock_conv_ps_kernel(const float* __restrict__ x,
                             const float* __restrict__ Wm,
                             const float* __restrict__ bias,
                             float* __restrict__ out)
{
  __shared__ __bf16 sW[OUT_CONV * LSTRIDE];   // W as bf16, [o][c], padded rows
  __shared__ __bf16 sX[TILE_PX * LSTRIDE];    // x tile transposed, [p][c]
  __shared__ float  sBias[OUT_CONV];

  const int tid  = (int)threadIdx.x;
  const int lane = tid & 31;
  const int wv   = tid >> 5;        // 0..11
  const int l16  = lane & 15;
  const int hi   = lane >> 4;       // 0 or 1

  const int bid   = (int)blockIdx.x;
  const int b     = bid / TILES_PB;
  const int tile  = bid - b * TILES_PB;
  const int pbase = tile * TILE_PX;

  // ---- stage W: fp32 -> bf16, row-major padded. thread -> half-row ----
  {
    const int o  = tid >> 1;                  // 0..191
    const int cb = (tid & 1) * 96;            // half-row base
    const float4* src = (const float4*)(Wm + o * IN_C + cb);
    __bf16* dst = &sW[o * LSTRIDE + cb];
    #pragma unroll
    for (int j = 0; j < 24; ++j) {
      float4 v = src[j];
      v4bf pk = { f2bf_rne(v.x), f2bf_rne(v.y), f2bf_rne(v.z), f2bf_rne(v.w) };
      *(v4bf*)(dst + j * 4) = pk;             // 8B-aligned ds_store_b64
    }
  }

  // ---- stage x tile transposed: wave wv owns channels wv*16..+15,
  //      32 lanes cover 128 contiguous pixels (float4 each) ----
  {
    const float4* xs = (const float4*)(x + b * (IN_C * NPIX) + pbase);
    const int p0 = lane * 4;
    #pragma unroll
    for (int j = 0; j < 16; ++j) {
      const int c = wv * 16 + j;
      float4 v = xs[c * (NPIX / 4) + lane];   // fully coalesced b128 load
      sX[(p0 + 0) * LSTRIDE + c] = f2bf_rne(v.x);
      sX[(p0 + 1) * LSTRIDE + c] = f2bf_rne(v.y);
      sX[(p0 + 2) * LSTRIDE + c] = f2bf_rne(v.z);
      sX[(p0 + 3) * LSTRIDE + c] = f2bf_rne(v.w);
    }
  }
  if (tid < OUT_CONV) sBias[tid] = bias[tid];
  __syncthreads();

  // ---- per-wave tile assignment: 2 M-tiles x 4 N-tiles ----
  const int mi = wv >> 1;           // 0..5  -> mtiles {2mi, 2mi+1}  (12 total)
  const int ni = wv & 1;            // 0..1  -> ntiles {4ni .. 4ni+3} (8 total)

  v8f acc[2][4];
  #pragma unroll
  for (int m = 0; m < 2; ++m)
    #pragma unroll
    for (int n = 0; n < 4; ++n)
      #pragma unroll
      for (int i = 0; i < 8; ++i) acc[m][n][i] = 0.0f;

  // ---- K loop: 6 x K32 chunks, 8 WMMAs each (48 total per wave) ----
  #pragma unroll
  for (int kk = 0; kk < 6; ++kk) {
    // A fragments (16-bit A layout: elems 0-7 = K{hi*8..}, 8-15 = K{16+hi*8..})
    const int ka = kk * 32 + hi * 8;
    v16bf a[2];
    #pragma unroll
    for (int m = 0; m < 2; ++m) {
      const int row = (mi * 2 + m) * 16 + l16;
      v8bf lo = *(const v8bf*)&sW[row * LSTRIDE + ka];
      v8bf hh = *(const v8bf*)&sW[row * LSTRIDE + ka + 16];
      a[m] = __builtin_shufflevector(lo, hh, 0,1,2,3,4,5,6,7,8,9,10,11,12,13,14,15);
    }
    // B fragments (lane = N = pixel; elems 0-7 = K{hi*8..}, 8-15 = K{16+hi*8..})
    const int kb = kk * 32 + hi * 8;
    #pragma unroll
    for (int n = 0; n < 4; ++n) {
      const int prow = (ni * 4 + n) * 16 + l16;
      v8bf lo = *(const v8bf*)&sX[prow * LSTRIDE + kb];
      v8bf hh = *(const v8bf*)&sX[prow * LSTRIDE + kb + 16];
      v16bf bb = __builtin_shufflevector(lo, hh, 0,1,2,3,4,5,6,7,8,9,10,11,12,13,14,15);
      #pragma unroll
      for (int m = 0; m < 2; ++m)
        acc[m][n] = __builtin_amdgcn_wmma_f32_16x16x32_bf16(
            false, a[m], false, bb, (short)0, acc[m][n], false, false);
    }
  }

  // ---- pixel-shuffle writeback (+bias), 64B-coalesced per 16 lanes ----
  int pixoff[4];
  #pragma unroll
  for (int n = 0; n < 4; ++n) {
    const unsigned p = (unsigned)(pbase + (ni * 4 + n) * 16 + l16);
    const unsigned h = p / 120u;
    const unsigned w = p - h * 120u;
    pixoff[n] = (int)(h * 960u + w);
  }
  const int obatch = b * (3 * 640 * 960);     // 1,843,200 per batch
  #pragma unroll
  for (int m = 0; m < 2; ++m) {
    #pragma unroll
    for (int v = 0; v < 8; ++v) {
      const int o   = (mi * 2 + m) * 16 + v + 8 * hi;  // C/D layout: M = v + 8*hi
      const int co  = o >> 6;
      const int rel = o & 63;
      const int chanoff = co * 614400 + (rel >> 3) * 76800 + (rel & 7) * 120;
      const float bv = sBias[o];
      #pragma unroll
      for (int n = 0; n < 4; ++n)
        out[obatch + chanoff + pixoff[n]] = acc[m][n][v] + bv;
    }
  }
}

extern "C" void kernel_launch(void* const* d_in, const int* in_sizes, int n_in,
                              void* d_out, int out_size, void* d_ws, size_t ws_size,
                              hipStream_t stream) {
  (void)in_sizes; (void)n_in; (void)d_ws; (void)ws_size; (void)out_size;
  const float* x    = (const float*)d_in[0];
  const float* Wm   = (const float*)d_in[1];
  const float* bias = (const float*)d_in[2];
  float* out        = (float*)d_out;

  dim3 grid(NB * TILES_PB);   // 16 * 75 = 1200 workgroups
  dim3 block(384);            // 12 waves
  subblock_conv_ps_kernel<<<grid, block, 0, stream>>>(x, Wm, bias, out);
}
